// MVAE_42425686950301
// MI455X (gfx1250) — matrix operand order repaired and checked by reading, compiled, not run on hardware
//
#include <hip/hip_runtime.h>
#include <cmath>

typedef __attribute__((ext_vector_type(16))) _Float16 v16h;
typedef __attribute__((ext_vector_type(8)))  float    v8f;

#define NTHREADS 512
#define NWAVES   16
#define NEXP     6
#define LO_SCALE 2048.0f
#define LO_INV   (1.0f / 2048.0f)

__device__ __forceinline__ int iclampmax(int v, int hi) { return v < hi ? v : hi; }

__device__ __forceinline__ float elu_f(float x) { return x > 0.0f ? x : expm1f(x); }

// A-matrix (16x32 f16) per CDNA5 ISA layout:
// lane L: row M = L&15; halves j=0..7  -> K = kbase + 8*(L>=16) + j
//          halves j=8..15 -> K = kbase + 16 + 8*(L>=16) + (j-8)
// Split each f32 weight into hi-f16 + (residual * 2048)-f16.
// Addresses are clamped (not values): padded-K columns always multiply a zero
// in B, and clamped rows are never written back, so loaded garbage is inert.
template<int M, int K>
__device__ __forceinline__ void load_A_split(const float* __restrict__ W,
                                             int row, int kbase, int hi,
                                             v16h& ah, v16h& al) {
    const float* Wr = W + (long)iclampmax(row, M - 1) * K;
    #pragma unroll
    for (int j = 0; j < 8; ++j) {
        float w0 = Wr[iclampmax(kbase + hi * 8 + j,      K - 1)];
        float w1 = Wr[iclampmax(kbase + 16 + hi * 8 + j, K - 1)];
        _Float16 h0 = (_Float16)w0, h1 = (_Float16)w1;
        ah[j]     = h0;
        ah[j + 8] = h1;
        al[j]     = (_Float16)((w0 - (float)h0) * LO_SCALE);
        al[j + 8] = (_Float16)((w1 - (float)h1) * LO_SCALE);
    }
}

// B-matrix (32x16 f16) broadcast: B[k][n] = x[k] for all n.
// lane L: halves j=0..15 -> K = kbase + 8*(L>=16) + (j&7) + 16*(j>>3)  (value N-independent).
// x must be zero-padded in LDS to a multiple of 32.
__device__ __forceinline__ void load_B_split(const float* __restrict__ xs,
                                             int kbase, int hi,
                                             v16h& bh, v16h& bl) {
    #pragma unroll
    for (int j = 0; j < 16; ++j) {
        float xv = xs[kbase + 8 * hi + (j & 7) + 16 * (j >> 3)];
        _Float16 h = (_Float16)xv;
        bh[j] = h;
        bl[j] = (_Float16)((xv - (float)h) * LO_SCALE);
    }
}

// One 16-row output tile of y = act(W @ x + b) with split-f16 precision recovery.
template<int ACT, int M, int K>
__device__ __forceinline__ void tile16(const float* __restrict__ W,
                                       const float* __restrict__ bias,
                                       const float* __restrict__ x,
                                       float* __restrict__ y,
                                       int rowbase, int m, int hi) {
    constexpr int nk = (K + 31) / 32;
    v8f c0 = {}, c1 = {}, c2 = {};
    for (int kc = 0; kc < nk; ++kc) {
        v16h ah, al, bh, bl;
        load_A_split<M, K>(W, rowbase + m, kc * 32, hi, ah, al);
        load_B_split(x, kc * 32, hi, bh, bl);
        c0 = __builtin_amdgcn_wmma_f32_16x16x32_f16(false, ah, false, bh, (short)0, c0, false, false);
        c1 = __builtin_amdgcn_wmma_f32_16x16x32_f16(false, al, false, bh, (short)0, c1, false, false);
        c2 = __builtin_amdgcn_wmma_f32_16x16x32_f16(false, ah, false, bl, (short)0, c2, false, false);
    }
    if (m == 0) {  // lanes 0 (rows +0..7) and 16 (rows +8..15) hold the result
        #pragma unroll
        for (int r = 0; r < 8; ++r) {
            int row = rowbase + r + hi * 8;
            if (row < M) {
                float v = c0[r] + (c1[r] + c2[r]) * LO_INV + bias[row];
                y[row] = ACT ? elu_f(v) : v;
            }
        }
    }
}

// y[M] = act(W[MxK] @ x[K] + bias[M]); all waves cooperate, 16 rows per tile.
template<int ACT, int M, int K>
__device__ void gemv16(const float* __restrict__ W, const float* __restrict__ bias,
                       const float* __restrict__ x, float* __restrict__ y) {
    const int lane = threadIdx.x & 31;
    const int wave = threadIdx.x >> 5;
    const int hi   = lane >> 4;
    const int m    = lane & 15;
    constexpr int ntiles = (M + 15) / 16;
    for (int t = wave; t < ntiles; t += NWAVES)
        tile16<ACT, M, K>(W, bias, x, y, t * 16, m, hi);
}

// Batched over NEXP experts: W is [E][M][K], bias [E][M]; x stride xstride
// (0 = shared input); y stride ystride with offset yoff (write after z-prefix).
template<int ACT, int M, int K>
__device__ void gemv16_exp(const float* __restrict__ W, const float* __restrict__ bias,
                           const float* __restrict__ x, int xstride,
                           float* __restrict__ y, int ystride, int yoff) {
    const int lane = threadIdx.x & 31;
    const int wave = threadIdx.x >> 5;
    const int hi   = lane >> 4;
    const int m    = lane & 15;
    constexpr int ntiles = (M + 15) / 16;
    for (int t = wave; t < NEXP * ntiles; t += NWAVES) {
        const int e  = t / ntiles;
        const int rt = t % ntiles;
        tile16<ACT, M, K>(W + (long)e * M * K, bias + e * M,
                          x + e * xstride, y + e * ystride + yoff,
                          rt * 16, m, hi);
    }
}

__global__ __launch_bounds__(NTHREADS)
void MVAE_42425686950301_kernel(
        const float* __restrict__ prev, const float* __restrict__ curr,
        const float* __restrict__ eps,
        const float* __restrict__ ew0, const float* __restrict__ eb0,
        const float* __restrict__ ew1, const float* __restrict__ eb1,
        const float* __restrict__ ew2, const float* __restrict__ eb2,
        const float* __restrict__ muw, const float* __restrict__ mub,
        const float* __restrict__ lvw, const float* __restrict__ lvb,
        const float* __restrict__ gw0, const float* __restrict__ gb0,
        const float* __restrict__ gw1, const float* __restrict__ gb1,
        const float* __restrict__ gw2, const float* __restrict__ gb2,
        const float* __restrict__ gw3, const float* __restrict__ gb3,
        const float* __restrict__ xw0, const float* __restrict__ xb0,
        const float* __restrict__ xw1, const float* __restrict__ xb1,
        const float* __restrict__ xw2, const float* __restrict__ xb2,
        const float* __restrict__ xw3, const float* __restrict__ xb3,
        float* __restrict__ out) {
    __shared__ float s_xin[128];          // cat(prev,curr)=124, zero-padded to 128
    __shared__ float s_h1[256];
    __shared__ float s_h2[256];
    __shared__ float s_mu[32];
    __shared__ float s_lv[32];
    __shared__ float s_gin[96];           // cat(z,prev)=94, zero-padded to 96
    __shared__ float s_gates[16];
    __shared__ float s_einA[NEXP][288];   // [0..31]=z, [32..287]=hidden
    __shared__ float s_einB[NEXP][288];
    __shared__ float s_oute[NEXP][64];

    const int tid = threadIdx.x;

    // --- stage encoder input (zero-padded) ---
    if (tid < 62) { s_xin[tid] = prev[tid]; s_xin[62 + tid] = curr[tid]; }
    if (tid >= 124 && tid < 128) s_xin[tid] = 0.0f;
    __syncthreads();

    // --- encoder ---
    gemv16<1, 256, 124>(ew0, eb0, s_xin, s_h1);
    __syncthreads();
    gemv16<1, 256, 256>(ew1, eb1, s_h1, s_h2);
    __syncthreads();
    gemv16<1, 256, 256>(ew2, eb2, s_h2, s_h1);
    __syncthreads();
    gemv16<0, 32, 256>(muw, mub, s_h1, s_mu);
    gemv16<0, 32, 256>(lvw, lvb, s_h1, s_lv);
    __syncthreads();

    // --- reparameterize; build g_in and per-expert z prefixes ---
    if (tid < 32) {
        float z = s_mu[tid] + eps[tid] * expf(0.5f * s_lv[tid]);
        s_gin[tid] = z;
        #pragma unroll
        for (int e = 0; e < NEXP; ++e) { s_einA[e][tid] = z; s_einB[e][tid] = z; }
    }
    if (tid < 62) s_gin[32 + tid] = prev[tid];
    if (tid >= 94 && tid < 96) s_gin[tid] = 0.0f;
    __syncthreads();

    // --- gate net + expert stack, layer by layer ---
    gemv16<1, 256, 94>(gw0, gb0, s_gin, s_h1);
    gemv16_exp<1, 256, 94>(xw0, xb0, s_gin, 0, &s_einA[0][0], 288, 32);
    __syncthreads();
    gemv16<1, 256, 256>(gw1, gb1, s_h1, s_h2);
    gemv16_exp<1, 256, 288>(xw1, xb1, &s_einA[0][0], 288, &s_einB[0][0], 288, 32);
    __syncthreads();
    gemv16<1, 256, 256>(gw2, gb2, s_h2, s_h1);
    gemv16_exp<1, 256, 288>(xw2, xb2, &s_einB[0][0], 288, &s_einA[0][0], 288, 32);
    __syncthreads();
    gemv16<0, 6, 256>(gw3, gb3, s_h1, s_gates);
    gemv16_exp<0, 62, 288>(xw3, xb3, &s_einA[0][0], 288, &s_oute[0][0], 64, 0);
    __syncthreads();

    // --- gated sum of experts; emit (decoded, mu, logvar) ---
    if (tid < 62) {
        float acc = 0.0f;
        #pragma unroll
        for (int e = 0; e < NEXP; ++e) acc += s_oute[e][tid] * s_gates[e];
        out[tid] = acc;
    }
    if (tid < 32) {
        out[62 + tid] = s_mu[tid];
        out[94 + tid] = s_lv[tid];
    }
}

extern "C" void kernel_launch(void* const* d_in, const int* in_sizes, int n_in,
                              void* d_out, int out_size, void* d_ws, size_t ws_size,
                              hipStream_t stream) {
    (void)in_sizes; (void)n_in; (void)out_size; (void)d_ws; (void)ws_size;
    const float* p[29];
    for (int i = 0; i < 29; ++i) p[i] = (const float*)d_in[i];
    MVAE_42425686950301_kernel<<<1, NTHREADS, 0, stream>>>(
        p[0],  p[1],  p[2],
        p[3],  p[4],  p[5],  p[6],  p[7],  p[8],
        p[9],  p[10], p[11], p[12],
        p[13], p[14], p[15], p[16], p[17], p[18], p[19], p[20],
        p[21], p[22], p[23], p[24], p[25], p[26], p[27], p[28],
        (float*)d_out);
}
